// OCAB_30356828848471
// MI455X (gfx1250) — matrix operand order repaired and checked by reading, not compiled, not run on hardware
//
#include <hip/hip_runtime.h>
#include <hip/hip_bf16.h>

// ---------------------------------------------------------------------------
// OCAB block (LN -> Haar DWT -> overlapped-window attention -> IDWT -> proj
// -> residual -> LN -> MLP -> residual) for MI455X (gfx1250, wave32, WMMA).
// All GEMM-like contractions use v_wmma_f32_16x16x32_f16 (f32 accumulate).
// Tile staging uses gfx1250 async global->LDS DMA (ASYNCcnt).
// ---------------------------------------------------------------------------

typedef __attribute__((ext_vector_type(16))) _Float16 v16h;
typedef __attribute__((ext_vector_type(8)))  _Float16 v8h;
typedef __attribute__((ext_vector_type(8)))  float    v8f;

#define DIMC   192
#define C4     768
#define HEADS  6
#define HD     32
#define HD4    128
#define WSZ    16
#define OWSZ   24
#define NK     576      // OWSZ*OWSZ
#define NQ     256      // WSZ*WSZ
#define H2     128
#define W2     128
#define NB     2
#define MLPH   384
#define NTOK2  32768    // NB*H2*W2
#define NTOK   131072   // NB*256*256

__device__ inline v8f zero8f(){ v8f r;
#pragma unroll
  for (int i=0;i<8;i++) r[i]=0.f; return r; }
__device__ inline v8h zero8h(){ v8h r;
#pragma unroll
  for (int i=0;i<8;i++) r[i]=(_Float16)0.f; return r; }
__device__ inline v16h comb16(v8h lo, v8h hi){ v16h r;
#pragma unroll
  for (int i=0;i<8;i++){ r[i]=lo[i]; r[8+i]=hi[i]; } return r; }

__device__ inline v8f wmma_f16f32(v16h a, v16h b, v8f c){
  return __builtin_amdgcn_wmma_f32_16x16x32_f16(false, a, false, b, (short)0, c,
                                                false, false);
}

// A fragment (16M x 32K, f16, row-major source): lane m=L&15, h=L>>4;
// K = [16h..16h+15] of row m.
__device__ inline v16h load_a_frag(const _Float16* base, int ld){
  int lane = threadIdx.x & 31; int m = lane & 15; int h = lane >> 4;
  const _Float16* p = base + m*ld + 16*h;
  return comb16(*(const v8h*)p, *(const v8h*)(p+8));
}
// B fragment (32K x 16N) loaded from B^T (N rows x K cols): lane n=L&15,
// h=L>>4; halves K = [16h..16h+15] of row n.
__device__ inline v16h load_b_frag(const _Float16* baseT, int ld){
  int lane = threadIdx.x & 31; int n = lane & 15; int h = lane >> 4;
  const _Float16* p = baseT + n*ld + 16*h;
  return comb16(*(const v8h*)p, *(const v8h*)(p+8));
}

// Wait for all of this wave's outstanding async global<->LDS transfers.
__device__ inline void async_wait(){
  asm volatile("s_wait_asynccnt 0x0" ::: "memory");
}

// Copy a rows x 32 f16 tile global -> LDS (LDS stride 40 halves) using the
// gfx1250 async-to-LDS DMA path: no VGPR round trip, tracked by ASYNCcnt.
// Caller must async_wait(); __syncthreads(); before consuming the tile.
__device__ inline void stage_tile(const _Float16* __restrict__ g, int ldg,
                                  _Float16* s, int rows){
  int elems = rows << 5;
  for (int idx = threadIdx.x*8; idx < elems; idx += blockDim.x*8){
    int r = idx >> 5; int c = idx & 31;
    unsigned lds = (unsigned)(size_t)(s + r*40 + c);        // LDS byte offset
    const _Float16* gp = g + (size_t)r*ldg + c;
    asm volatile("global_load_async_to_lds_b128 %0, %1, off"
                 :: "v"(lds), "v"(gp) : "memory");
  }
}

// 192-thread (6-wave) block sum.
__device__ inline float block_sum192(float v, float* sred){
  __syncthreads();
#pragma unroll
  for (int o=16;o>0;o>>=1) v += __shfl_down(v, o, 32);
  int wi = threadIdx.x >> 5;
  if ((threadIdx.x & 31) == 0) sred[wi] = v;
  __syncthreads();
  float s = 0.f;
#pragma unroll
  for (int i=0;i<6;i++) s += sred[i];
  return s;
}

// ---------------------------------------------------------------------------
// K0: weight f32 (K x N) -> f16 transposed (N x K)
// ---------------------------------------------------------------------------
__global__ void wconv(const float* __restrict__ w, _Float16* __restrict__ wT,
                      int K, int N){
  int e = blockIdx.x*256 + threadIdx.x;
  if (e >= K*N) return;
  int k = e / N, n = e - k*N;
  wT[(size_t)n*K + k] = (_Float16)w[e];
}

// ---------------------------------------------------------------------------
// K1: LayerNorm + Haar DWT. One block per DWT token (b,y,x), 192 threads.
// ---------------------------------------------------------------------------
__global__ __launch_bounds__(192)
void ln_dwt(const float* __restrict__ x, const float* __restrict__ g,
            const float* __restrict__ be, _Float16* __restrict__ xd){
  __shared__ float sred[8];
  int t = blockIdx.x;
  int b = t >> 14; int rem = t & 16383;
  int y = rem >> 7; int xw = rem & 127;
  int f = threadIdx.x;
  float gf = g[f], bf = be[f];
  float vals[4], n[4];
#pragma unroll
  for (int i=0;i<4;i++){
    int yy = 2*y + (i>>1), xx = 2*xw + (i&1);
    vals[i] = x[((size_t)b*65536 + yy*256 + xx)*DIMC + f];
  }
#pragma unroll
  for (int i=0;i<4;i++){
    float s  = block_sum192(vals[i], sred);
    float mu = s * (1.f/192.f);
    float d  = vals[i] - mu;
    float s2 = block_sum192(d*d, sred);
    n[i] = d * rsqrtf(s2*(1.f/192.f) + 1e-5f) * gf + bf;
  }
  float a=n[0], bb=n[1], cc=n[2], dd=n[3];
  size_t base = (size_t)t*C4 + f;
  xd[base      ] = (_Float16)(0.5f*( a+bb+cc+dd));
  xd[base+DIMC ] = (_Float16)(0.5f*(-a-bb+cc+dd));
  xd[base+2*DIMC] = (_Float16)(0.5f*(-a+bb-cc+dd));
  xd[base+3*DIMC] = (_Float16)(0.5f*( a-bb-cc+dd));
}

// ---------------------------------------------------------------------------
// K2: generic f16 GEMM  C = A(MxK) * B(KxN) + bias.  B passed transposed.
// Output cols < split -> C0 (scaled by scale0, ld=split); else C1.
// Tiles: 128x128x32. 8 waves: 2 m-tiles x 4 n-tiles each.
// ---------------------------------------------------------------------------
__global__ __launch_bounds__(256)
void gemm_f16(const _Float16* __restrict__ A, const _Float16* __restrict__ BT,
              const float* __restrict__ bias,
              _Float16* __restrict__ C0, _Float16* __restrict__ C1,
              int split, float scale0, int M, int N, int K){
  __shared__ _Float16 sA[128*40];
  __shared__ _Float16 sB[128*40];
  int m0 = blockIdx.x << 7, n0 = blockIdx.y << 7;
  int w = threadIdx.x >> 5, lane = threadIdx.x & 31;
  int mw = w & 3, nw = w >> 2;
  int hl = lane >> 4, nl = lane & 15;
  v8f acc[2][4];
#pragma unroll
  for (int i=0;i<2;i++)
#pragma unroll
    for (int j=0;j<4;j++) acc[i][j] = zero8f();
  for (int kb = 0; kb < K; kb += 32){
    stage_tile(A  + (size_t)m0*K + kb, K, sA, 128);
    stage_tile(BT + (size_t)n0*K + kb, K, sB, 128);
    async_wait();
    __syncthreads();
    v16h af[2], bf[4];
#pragma unroll
    for (int i=0;i<2;i++) af[i] = load_a_frag(sA + (mw*32 + i*16)*40, 40);
#pragma unroll
    for (int j=0;j<4;j++) bf[j] = load_b_frag(sB + (nw*64 + j*16)*40, 40);
#pragma unroll
    for (int i=0;i<2;i++)
#pragma unroll
      for (int j=0;j<4;j++) acc[i][j] = wmma_f16f32(af[i], bf[j], acc[i][j]);
    __syncthreads();
  }
#pragma unroll
  for (int i=0;i<2;i++)
#pragma unroll
  for (int j=0;j<4;j++)
#pragma unroll
  for (int r=0;r<8;r++){
    int m = m0 + mw*32 + i*16 + r + 8*hl;
    int n = n0 + nw*64 + j*16 + nl;
    float v = acc[i][j][r] + bias[n];
    if (n < split) C0[(size_t)m*split + n] = (_Float16)(v*scale0);
    else           C1[(size_t)m*(N-split) + (n-split)] = (_Float16)v;
  }
}

// ---------------------------------------------------------------------------
// K3: overlapped-window attention (flash-style online softmax).
// grid = 128 windows * 6 heads * 2 q-halves. Each wave owns 16 q rows.
// K-dim of scores == head_dim == 32 -> one WMMA per 16x16 score tile.
// k/v rows gathered with pad=4 overlap bounds check (OOB -> 0, matching ref).
// ---------------------------------------------------------------------------
__global__ __launch_bounds__(256)
void attn_kernel(const _Float16* __restrict__ qg, const _Float16* __restrict__ kg,
                 const _Float16* __restrict__ vg, const int* __restrict__ rpi,
                 const float* __restrict__ rpb, _Float16* __restrict__ ao){
  __shared__ _Float16 vT[128*40];      // v chunk transposed: [feat 0..127][tok 0..31]
  __shared__ _Float16 pbuf[8*16*40];   // per-wave P tile (16x32)
  int bx = blockIdx.x;
  int wid = bx / 12; int rem = bx - wid*12;
  int head = rem >> 1; int qh = rem & 1;
  int bb = wid >> 6; int wy = (wid >> 3) & 7; int wx = wid & 7;
  int w = threadIdx.x >> 5, lane = threadIdx.x & 31;
  int nl = lane & 15, hl = lane >> 4;
  _Float16* pw = pbuf + w*16*40;
  int q0 = (qh << 7) + (w << 4);
  v16h qf;
  { int qi = q0 + nl;
    int y = (wy<<4) + (qi>>4), x = (wx<<4) + (qi&15);
    const _Float16* p = qg + ((size_t)(((bb<<7)+y)*128 + x))*DIMC + head*HD + 16*hl;
    qf = comb16(*(const v8h*)p, *(const v8h*)(p+8));
  }
  v8f oacc[8];
#pragma unroll
  for (int j=0;j<8;j++) oacc[j] = zero8f();
  float mi[8], li[8];
#pragma unroll
  for (int r=0;r<8;r++){ mi[r] = -3.0e38f; li[r] = 0.f; }

  for (int kb = 0; kb < NK; kb += 32){
    { // stage v chunk transposed (32 tokens x 128 feats)
      int tok = kb + (threadIdx.x >> 3);
      int fb  = (threadIdx.x & 7) << 4;
      int wr = tok / OWSZ; int wc = tok - wr*OWSZ;
      int y = (wy<<4) - 4 + wr, x = (wx<<4) - 4 + wc;
      v8h d0 = zero8h(), d1 = zero8h();
      if (y >= 0 && y < H2 && x >= 0 && x < W2){
        const _Float16* p = vg + ((size_t)(((bb<<7)+y)*128 + x))*C4 + head*HD4 + fb;
        d0 = *(const v8h*)p; d1 = *(const v8h*)(p+8);
      }
      int tl = tok - kb;
#pragma unroll
      for (int i=0;i<8;i++){ vT[(fb+i)*40 + tl]   = d0[i];
                             vT[(fb+8+i)*40 + tl] = d1[i]; }
    }
    __syncthreads();
    // two 16x16 score tiles: S = q (16x32) * k^T
    v8f s01[2];
#pragma unroll
    for (int hf=0; hf<2; hf++){
      int ki = kb + (hf<<4) + nl;
      int wr = ki / OWSZ; int wc = ki - wr*OWSZ;
      int y = (wy<<4) - 4 + wr, x = (wx<<4) - 4 + wc;
      v8h k0 = zero8h(), k1 = zero8h();
      if (y >= 0 && y < H2 && x >= 0 && x < W2){
        const _Float16* p = kg + ((size_t)(((bb<<7)+y)*128 + x))*DIMC + head*HD + 16*hl;
        k0 = *(const v8h*)p; k1 = *(const v8h*)(p+8);
      }
      s01[hf] = wmma_f16f32(qf, comb16(k0, k1), zero8f());
    }
    // bias + online softmax per row (rows r+8*hl live in 16-lane half-waves)
#pragma unroll
    for (int r=0;r<8;r++){
      int qi = q0 + r + 8*hl;
      float b0 = rpb[rpi[qi*NK + kb + nl]*HEADS + head];
      float b1 = rpb[rpi[qi*NK + kb + 16 + nl]*HEADS + head];
      float v0 = s01[0][r] + b0;
      float v1 = s01[1][r] + b1;
      float rm = fmaxf(v0, v1);
#pragma unroll
      for (int o=1;o<16;o<<=1) rm = fmaxf(rm, __shfl_xor(rm, o, 32));
      float mn = fmaxf(mi[r], rm);
      float alpha = __expf(mi[r] - mn);
      float p0 = __expf(v0 - mn), p1 = __expf(v1 - mn);
      float rs = p0 + p1;
#pragma unroll
      for (int o=1;o<16;o<<=1) rs += __shfl_xor(rs, o, 32);
      li[r] = li[r]*alpha + rs; mi[r] = mn;
#pragma unroll
      for (int j=0;j<8;j++) oacc[j][r] *= alpha;
      pw[(r + 8*hl)*40 + nl]      = (_Float16)p0;
      pw[(r + 8*hl)*40 + 16 + nl] = (_Float16)p1;
    }
    __syncthreads();
    // out += P (16x32) * v_chunk (32x128)
    v16h pf = load_a_frag(pw, 40);
#pragma unroll
    for (int j=0;j<8;j++){
      v16h vf = load_b_frag(vT + (j<<4)*40, 40);
      oacc[j] = wmma_f16f32(pf, vf, oacc[j]);
    }
    __syncthreads();
  }
  // normalize + window-reverse store
#pragma unroll
  for (int r=0;r<8;r++){
    int qi = q0 + r + 8*hl;
    int y = (wy<<4) + (qi>>4), x = (wx<<4) + (qi&15);
    size_t base = ((size_t)(((bb<<7)+y)*128 + x))*C4 + head*HD4;
    float inv = 1.f/li[r];
#pragma unroll
    for (int j=0;j<8;j++) ao[base + (j<<4) + nl] = (_Float16)(oacc[j][r]*inv);
  }
}

// ---------------------------------------------------------------------------
// K4: IDWT (fused into A-tile load) + proj GEMM + bias + shortcut -> d_out(x1)
// ---------------------------------------------------------------------------
__global__ __launch_bounds__(256)
void idwt_proj(const _Float16* __restrict__ ao, const _Float16* __restrict__ projT,
               const float* __restrict__ pb, const float* __restrict__ xin,
               float* __restrict__ out){
  __shared__ _Float16 sA[128*40];
  __shared__ _Float16 sB[64*40];
  int m0 = blockIdx.x << 7, n0 = blockIdx.y << 6;
  int w = threadIdx.x >> 5, lane = threadIdx.x & 31;
  int mw = w & 3, nw = w >> 2, hl = lane >> 4, nl = lane & 15;
  v8f acc[2][2];
#pragma unroll
  for (int i=0;i<2;i++)
#pragma unroll
    for (int j=0;j<2;j++) acc[i][j] = zero8f();

  int row = threadIdx.x >> 1; int cb = (threadIdx.x & 1) << 4;
  size_t tt = (size_t)m0 + row;
  int b = (int)(tt >> 16); int pix = (int)(tt & 65535);
  int Y = pix >> 8, X = pix & 255;
  int y = Y >> 1, x = X >> 1;
  float slh = (Y & 1) ? 1.f : -1.f;
  float shl = (X & 1) ? 1.f : -1.f;
  float shh = slh*shl;
  const _Float16* arow = ao + ((size_t)(((b<<7)+y)*128 + x))*C4;

  for (int kb=0; kb<DIMC; kb+=32){
    stage_tile(projT + (size_t)n0*DIMC + kb, DIMC, sB, 64);
#pragma unroll
    for (int c8=0; c8<16; c8+=8){
      const _Float16* p = arow + kb + cb + c8;
      v8h ll = *(const v8h*)p;
      v8h lh = *(const v8h*)(p+DIMC);
      v8h hv = *(const v8h*)(p+2*DIMC);
      v8h hh = *(const v8h*)(p+3*DIMC);
#pragma unroll
      for (int i=0;i<8;i++){
        float vv = 0.5f*((float)ll[i] + slh*(float)lh[i]
                         + shl*(float)hv[i] + shh*(float)hh[i]);
        sA[row*40 + cb + c8 + i] = (_Float16)vv;
      }
    }
    async_wait();
    __syncthreads();
    v16h af[2];
#pragma unroll
    for (int i=0;i<2;i++) af[i] = load_a_frag(sA + (mw*32 + i*16)*40, 40);
#pragma unroll
    for (int i=0;i<2;i++)
#pragma unroll
      for (int j=0;j<2;j++){
        v16h bf = load_b_frag(sB + (nw*32 + j*16)*40, 40);
        acc[i][j] = wmma_f16f32(af[i], bf, acc[i][j]);
      }
    __syncthreads();
  }
#pragma unroll
  for (int i=0;i<2;i++)
#pragma unroll
  for (int j=0;j<2;j++)
#pragma unroll
  for (int r=0;r<8;r++){
    int m = m0 + mw*32 + i*16 + r + 8*hl;
    int n = n0 + nw*32 + j*16 + nl;
    out[(size_t)m*DIMC + n] = acc[i][j][r] + pb[n] + xin[(size_t)m*DIMC + n];
  }
}

// ---------------------------------------------------------------------------
// K5: LayerNorm2: x1 (d_out) -> hdn f16
// ---------------------------------------------------------------------------
__global__ __launch_bounds__(192)
void ln2_kernel(const float* __restrict__ x1, const float* __restrict__ g,
                const float* __restrict__ be, _Float16* __restrict__ hdn){
  __shared__ float sred[8];
  size_t t = blockIdx.x; int f = threadIdx.x;
  float v = x1[t*DIMC + f];
  float s  = block_sum192(v, sred);
  float mu = s * (1.f/192.f);
  float d  = v - mu;
  float s2 = block_sum192(d*d, sred);
  hdn[t*DIMC + f] = (_Float16)(d * rsqrtf(s2*(1.f/192.f) + 1e-5f) * g[f] + be[f]);
}

// ---------------------------------------------------------------------------
// K6: fused MLP: out += gelu(hdn@fc1 + b1) @ fc2 + b2. hid kept in 64-wide
// chunks in LDS; fc2 result accumulated in registers (no global hid buffer).
// ---------------------------------------------------------------------------
__global__ __launch_bounds__(256)
void mlp_kernel(const _Float16* __restrict__ hdn, const _Float16* __restrict__ fc1T,
                const float* __restrict__ b1, const _Float16* __restrict__ fc2T,
                const float* __restrict__ b2, float* __restrict__ out){
  __shared__ _Float16 hch[128*72];
  __shared__ _Float16 sB[192*40];
  int m0 = blockIdx.x << 7;
  int w = threadIdx.x >> 5, lane = threadIdx.x & 31;
  int mw = w & 3, nw = w >> 2, hl = lane >> 4, nl = lane & 15;
  v8f oacc[2][6];
#pragma unroll
  for (int i=0;i<2;i++)
#pragma unroll
    for (int j=0;j<6;j++) oacc[i][j] = zero8f();

  for (int hc=0; hc<MLPH; hc+=64){
    v8f c1[2][2];
#pragma unroll
    for (int i=0;i<2;i++)
#pragma unroll
      for (int j=0;j<2;j++) c1[i][j] = zero8f();
    for (int kb=0; kb<DIMC; kb+=32){
      stage_tile(fc1T + (size_t)hc*DIMC + kb, DIMC, sB, 64);
      async_wait();
      __syncthreads();
#pragma unroll
      for (int i=0;i<2;i++){
        v16h af = load_a_frag(hdn + (size_t)(m0 + mw*32 + i*16)*DIMC + kb, DIMC);
#pragma unroll
        for (int j=0;j<2;j++){
          v16h bf = load_b_frag(sB + (nw*32 + j*16)*40, 40);
          c1[i][j] = wmma_f16f32(af, bf, c1[i][j]);
        }
      }
      __syncthreads();
    }
    // bias + exact GELU -> LDS hid chunk
#pragma unroll
    for (int i=0;i<2;i++)
#pragma unroll
    for (int j=0;j<2;j++)
#pragma unroll
    for (int r=0;r<8;r++){
      int rr = mw*32 + i*16 + r + 8*hl;
      int cc = nw*32 + j*16 + nl;
      float vv = c1[i][j][r] + b1[hc + cc];
      vv = 0.5f*vv*(1.f + erff(vv*0.70710678118654752f));
      hch[rr*72 + cc] = (_Float16)vv;
    }
    __syncthreads();
    for (int k2=0; k2<64; k2+=32){
      stage_tile(fc2T + hc + k2, MLPH, sB, 192);
      async_wait();
      __syncthreads();
#pragma unroll
      for (int i=0;i<2;i++){
        v16h af = load_a_frag(hch + (mw*32 + i*16)*72 + k2, 72);
#pragma unroll
        for (int j=0;j<6;j++){
          v16h bf = load_b_frag(sB + (nw*96 + j*16)*40, 40);
          oacc[i][j] = wmma_f16f32(af, bf, oacc[i][j]);
        }
      }
      __syncthreads();
    }
  }
#pragma unroll
  for (int i=0;i<2;i++)
#pragma unroll
  for (int j=0;j<6;j++)
#pragma unroll
  for (int r=0;r<8;r++){
    int m = m0 + mw*32 + i*16 + r + 8*hl;
    int n = nw*96 + j*16 + nl;
    out[(size_t)m*DIMC + n] += oacc[i][j][r] + b2[n];
  }
}

// ---------------------------------------------------------------------------
extern "C" void kernel_launch(void* const* d_in, const int* in_sizes, int n_in,
                              void* d_out, int out_size, void* d_ws, size_t ws_size,
                              hipStream_t stream) {
  (void)in_sizes; (void)n_in; (void)out_size; (void)ws_size;
  const float* x      = (const float*)d_in[0];
  const int*   rpi    = (const int*)d_in[1];
  const float* n1g    = (const float*)d_in[4];
  const float* n1b    = (const float*)d_in[5];
  const float* qk_w   = (const float*)d_in[6];
  const float* qk_b   = (const float*)d_in[7];
  const float* v_w    = (const float*)d_in[8];
  const float* v_b    = (const float*)d_in[9];
  const float* rpb    = (const float*)d_in[10];
  const float* proj_w = (const float*)d_in[11];
  const float* proj_b = (const float*)d_in[12];
  const float* n2g    = (const float*)d_in[13];
  const float* n2b    = (const float*)d_in[14];
  const float* fc1_w  = (const float*)d_in[15];
  const float* fc1_b  = (const float*)d_in[16];
  const float* fc2_w  = (const float*)d_in[17];
  const float* fc2_b  = (const float*)d_in[18];
  float* out = (float*)d_out;

  char* wsb = (char*)d_ws;
  _Float16* xd  = (_Float16*)wsb;                        // 32768*768 f16 (48MB)
  _Float16* ao  = xd;                                    // alias: xd dead after GEMMs
  _Float16* qf  = (_Float16*)(wsb + 50331648);           // 32768*192
  _Float16* kf  = qf + (size_t)NTOK2*DIMC;               // 32768*192
  _Float16* vf  = kf + (size_t)NTOK2*DIMC;               // 32768*768
  _Float16* hdn = qf;                                    // alias: q/k/v dead after attn
  _Float16* qkT = vf + (size_t)NTOK2*C4;                 // weights (f16, transposed)
  _Float16* vT  = qkT + (size_t)384*C4;
  _Float16* pjT = vT  + (size_t)C4*C4;
  _Float16* f1T = pjT + (size_t)DIMC*DIMC;
  _Float16* f2T = f1T + (size_t)MLPH*DIMC;

  const float scale = 0.17677669529663687f;  // 1/sqrt(32)

  wconv<<<(C4*384 + 255)/256, 256, 0, stream>>>(qk_w, qkT, C4, 384);
  wconv<<<(C4*C4  + 255)/256, 256, 0, stream>>>(v_w,  vT,  C4, C4);
  wconv<<<(DIMC*DIMC + 255)/256, 256, 0, stream>>>(proj_w, pjT, DIMC, DIMC);
  wconv<<<(DIMC*MLPH + 255)/256, 256, 0, stream>>>(fc1_w, f1T, DIMC, MLPH);
  wconv<<<(MLPH*DIMC + 255)/256, 256, 0, stream>>>(fc2_w, f2T, MLPH, DIMC);

  ln_dwt<<<NTOK2, 192, 0, stream>>>(x, n1g, n1b, xd);

  gemm_f16<<<dim3(NTOK2/128, 3), 256, 0, stream>>>(xd, qkT, qk_b, qf, kf,
                                                   DIMC, scale, NTOK2, 384, C4);
  gemm_f16<<<dim3(NTOK2/128, 6), 256, 0, stream>>>(xd, vT, v_b, vf, vf,
                                                   C4, 1.0f, NTOK2, C4, C4);

  attn_kernel<<<128*HEADS*2, 256, 0, stream>>>(qf, kf, vf, rpi, rpb, ao);

  idwt_proj<<<dim3(NTOK/128, 3), 256, 0, stream>>>(ao, pjT, proj_b, x, out);

  ln2_kernel<<<NTOK, 192, 0, stream>>>(out, n2g, n2b, hdn);

  mlp_kernel<<<NTOK/128, 256, 0, stream>>>(hdn, f1T, fc1_b, f2T, fc2_b, out);
}
